// VectorQuantizer1D_52493090291935
// MI455X (gfx1250) — matrix operand order repaired and checked by reading, not compiled, not run on hardware
//
#include <hip/hip_runtime.h>

// ---------------------------------------------------------------------------
// VQ-VAE vector quantizer for MI455X (gfx1250, wave32, WMMA)
//   z_e : [32,1024,256] f32   -> N = 32768 rows, D = 256
//   emb : [8192,256]    f32   -> K = 8192 codes
// Outputs (concatenated f32): z_q [N*D], indices-as-float [N], vq_loss [1]
//
// Score matmul (N x K x D) uses split-bf16 (hi/lo) V_WMMA_F32_16X16X32_BF16:
//   dot ~= ah*bh + ah*bl + al*bh   (~fp32 accuracy, 24 wmma per 16x16xD tile)
// ILP: 2 code tiles/iter x 3 partial-product accumulators = 6 independent
// WMMA chains, covering the 5-slot bf16 WMMA->WMMA RAW hazard window.
// ---------------------------------------------------------------------------

#define VQ_D      256
#define VQ_K      8192
#define VQ_N      32768
#define VQ_BETA   0.25f

typedef __attribute__((ext_vector_type(16))) __bf16 v16bf;
typedef __attribute__((ext_vector_type(8)))  float  v8f;

union Frag16 {
    v16bf          bf;
    unsigned short u[16];
};

// Round-to-nearest-even f32 -> bf16 split: f ~= hi + lo (both bf16).
__device__ __forceinline__ void bf16_split(float f, unsigned short& hi, unsigned short& lo) {
    unsigned int u  = __float_as_uint(f);
    unsigned int r  = u + 0x7FFFu + ((u >> 16) & 1u);
    hi = (unsigned short)(r >> 16);
    float fh  = __uint_as_float(((unsigned int)hi) << 16);
    float res = f - fh;
    unsigned int u2 = __float_as_uint(res);
    unsigned int r2 = u2 + 0x7FFFu + ((u2 >> 16) & 1u);
    lo = (unsigned short)(r2 >> 16);
}

__device__ __forceinline__ v8f wmma_bf16(v16bf a, v16bf b, v8f c) {
    return __builtin_amdgcn_wmma_f32_16x16x32_bf16(
        false, a, false, b, (short)0, c, false, false);
}

// ---------------------------------------------------------------------------
// Kernel 1: prep. One block per code row: emit bf16 hi/lo copies of emb,
// reduce ||e||^2, zero the loss accumulator.
// ---------------------------------------------------------------------------
__global__ __launch_bounds__(256) void vq_prep_kernel(
    const float* __restrict__ emb,
    unsigned short* __restrict__ emb_hi,
    unsigned short* __restrict__ emb_lo,
    float* __restrict__ e2,
    float* __restrict__ loss_acc)
{
    const int k = blockIdx.x;
    const int d = threadIdx.x;
    const size_t off = (size_t)k * VQ_D + d;

    float f = emb[off];
    unsigned short h, l;
    bf16_split(f, h, l);
    emb_hi[off] = h;
    emb_lo[off] = l;

    float s = f * f;
    #pragma unroll
    for (int o = 16; o >= 1; o >>= 1) s += __shfl_xor(s, o, 32);

    __shared__ float red[8];
    const int lane = threadIdx.x & 31;
    const int wv   = threadIdx.x >> 5;
    if (lane == 0) red[wv] = s;
    __syncthreads();
    if (threadIdx.x == 0) {
        float tot = 0.f;
        #pragma unroll
        for (int i = 0; i < 8; ++i) tot += red[i];
        e2[k] = tot;
        if (k == 0) *loss_acc = 0.0f;   // fresh accumulator every call
    }
}

// ---------------------------------------------------------------------------
// Kernel 2: argmin over codes via WMMA. Block = 8 waves, wave = 16 rows.
// A frags (z rows, split bf16) live in VGPRs for the whole code sweep.
// Per iteration: 2 code tiles (32 codes), 8 K-chunks, 6 independent wmma
// chains; running per-lane min; final 16-lane butterfly + index tie-break.
// ---------------------------------------------------------------------------
__global__ __launch_bounds__(256) void vq_argmin_kernel(
    const float* __restrict__ z_e,
    const unsigned short* __restrict__ emb_hi,
    const unsigned short* __restrict__ emb_lo,
    const float* __restrict__ e2,
    int*   __restrict__ idx_out,
    float* __restrict__ idx_f_out)
{
    const int lane    = threadIdx.x & 31;
    const int wave    = threadIdx.x >> 5;
    const int rowBase = blockIdx.x * 128 + wave * 16;
    const int half    = lane >> 4;   // K-half selector (ISA A/B lane-group layout)
    const int sub     = lane & 15;   // row index (A) / code column (B)

    // ---- preload A fragments: row (rowBase+sub), split into bf16 hi/lo ----
    // A 16x32 bf16 layout: lanes 0-15 hold K {0..15}, lanes 16-31 hold
    // K {16..31} of each 32-chunk (contiguous 16-element run, matches B).
    v16bf a_hi[8], a_lo[8];
    {
        const float* zrow = z_e + (size_t)(rowBase + sub) * VQ_D;
        #pragma unroll
        for (int t = 0; t < 8; ++t) {
            const int d0 = t * 32 + half * 16;
            Frag16 h, l;
            #pragma unroll
            for (int j = 0; j < 8; ++j) {
                bf16_split(zrow[d0 + j],      h.u[j],     l.u[j]);
                bf16_split(zrow[d0 + 8 + j],  h.u[8 + j], l.u[8 + j]);
            }
            a_hi[t] = h.bf;
            a_lo[t] = l.bf;
        }
    }

    float minval[8];
    int   minidx[8];
    #pragma unroll
    for (int v = 0; v < 8; ++v) { minval[v] = 3.4e38f; minidx[v] = 0; }

    // ---- sweep code tiles, 2 per iteration (32 codes) ----
    for (int ct = 0; ct < VQ_K / 32; ++ct) {
        const int code0 = ct * 32 + sub;
        const int code1 = code0 + 16;
        // B 32x16 bf16 layout: lane holds column `sub`, K-run of 16 starting
        // at half*16 within each 32-chunk -> one contiguous 32-byte load.
        const v16bf* bh0 = (const v16bf*)(emb_hi + (size_t)code0 * VQ_D + half * 16);
        const v16bf* bl0 = (const v16bf*)(emb_lo + (size_t)code0 * VQ_D + half * 16);
        const v16bf* bh1 = (const v16bf*)(emb_hi + (size_t)code1 * VQ_D + half * 16);
        const v16bf* bl1 = (const v16bf*)(emb_lo + (size_t)code1 * VQ_D + half * 16);

        v8f a00 = {}, a01 = {}, a02 = {};   // tile 0: hh, hl, lh partials
        v8f a10 = {}, a11 = {}, a12 = {};   // tile 1: hh, hl, lh partials
        #pragma unroll
        for (int t = 0; t < 8; ++t) {
            const v16bf bhv0 = bh0[2 * t];  // chunk t starts 32 elems apart
            const v16bf blv0 = bl0[2 * t];
            const v16bf bhv1 = bh1[2 * t];
            const v16bf blv1 = bl1[2 * t];
            // 6 independent accumulator chains -> dependent wmma 6 slots apart
            a00 = wmma_bf16(a_hi[t], bhv0, a00);
            a10 = wmma_bf16(a_hi[t], bhv1, a10);
            a01 = wmma_bf16(a_hi[t], blv0, a01);
            a11 = wmma_bf16(a_hi[t], blv1, a11);
            a02 = wmma_bf16(a_lo[t], bhv0, a02);
            a12 = wmma_bf16(a_lo[t], bhv1, a12);
        }

        const float e2v0 = e2[code0];
        const float e2v1 = e2[code1];
        #pragma unroll
        for (int v = 0; v < 8; ++v) {
            // ||z||^2 omitted (constant per row, drops out of argmin)
            const float d0 = e2v0 - 2.0f * (a00[v] + a01[v] + a02[v]);
            if (d0 < minval[v]) { minval[v] = d0; minidx[v] = code0; }
            const float d1 = e2v1 - 2.0f * (a10[v] + a11[v] + a12[v]);
            if (d1 < minval[v]) { minval[v] = d1; minidx[v] = code1; }
        }
    }

    // ---- butterfly min-reduce across the 16 lanes of each half-group ----
    #pragma unroll
    for (int o = 1; o < 16; o <<= 1) {
        #pragma unroll
        for (int v = 0; v < 8; ++v) {
            const float ov = __shfl_xor(minval[v], o, 32);
            const int   oi = __shfl_xor(minidx[v], o, 32);
            if (ov < minval[v] || (ov == minval[v] && oi < minidx[v])) {
                minval[v] = ov;
                minidx[v] = oi;
            }
        }
    }

    // C/D layout: lanes 0-15 carry rows v (VGPR v), lanes 16-31 rows v+8.
    if (sub == 0) {
        #pragma unroll
        for (int v = 0; v < 8; ++v) {
            const int row = rowBase + half * 8 + v;
            idx_out[row]   = minidx[v];
            idx_f_out[row] = (float)minidx[v];
        }
    }
}

// ---------------------------------------------------------------------------
// Kernel 3: gather z_q (fp32 emb rows) + squared-error reduction.
// ---------------------------------------------------------------------------
__global__ __launch_bounds__(256) void vq_gather_loss_kernel(
    const float* __restrict__ z_e,
    const float* __restrict__ emb,
    const int*   __restrict__ idx,
    float* __restrict__ out_zq,
    float* __restrict__ loss_acc)
{
    const int row = blockIdx.x;
    const int d   = threadIdx.x;
    const int k   = idx[row];

    const float q = emb[(size_t)k * VQ_D + d];
    const float z = z_e[(size_t)row * VQ_D + d];
    out_zq[(size_t)row * VQ_D + d] = q;

    const float diff = q - z;
    float s = diff * diff;
    #pragma unroll
    for (int o = 16; o >= 1; o >>= 1) s += __shfl_xor(s, o, 32);

    __shared__ float red[8];
    const int lane = threadIdx.x & 31;
    const int wv   = threadIdx.x >> 5;
    if (lane == 0) red[wv] = s;
    __syncthreads();
    if (threadIdx.x == 0) {
        float tot = 0.f;
        #pragma unroll
        for (int i = 0; i < 8; ++i) tot += red[i];
        atomicAdd(loss_acc, tot);
    }
}

// ---------------------------------------------------------------------------
// Kernel 4: finalize vq_loss = (1 + beta) * SSE / (N*D).
// ---------------------------------------------------------------------------
__global__ void vq_finalize_kernel(const float* __restrict__ loss_acc,
                                   float* __restrict__ out_loss)
{
    *out_loss = (1.0f + VQ_BETA) * (*loss_acc) / (float)((size_t)VQ_N * VQ_D);
}

// ---------------------------------------------------------------------------
extern "C" void kernel_launch(void* const* d_in, const int* in_sizes, int n_in,
                              void* d_out, int out_size, void* d_ws, size_t ws_size,
                              hipStream_t stream)
{
    const float* z_e = (const float*)d_in[0];   // 32*1024*256 f32
    const float* emb = (const float*)d_in[1];   // 8192*256   f32

    float* out      = (float*)d_out;
    float* out_zq   = out;                                   // [N*D]
    float* out_idx  = out + (size_t)VQ_N * VQ_D;             // [N]
    float* out_loss = out + (size_t)VQ_N * VQ_D + VQ_N;      // [1]

    char* ws = (char*)d_ws;
    unsigned short* emb_hi = (unsigned short*)ws;                               // 4 MB
    unsigned short* emb_lo = (unsigned short*)(ws + (size_t)VQ_K * VQ_D * 2);   // 4 MB
    float* e2       = (float*)(ws + (size_t)VQ_K * VQ_D * 4);                   // 32 KB
    int*   idx      = (int*)  (ws + (size_t)VQ_K * VQ_D * 4 + (size_t)VQ_K * 4);
    float* loss_acc = (float*)(ws + (size_t)VQ_K * VQ_D * 4 + (size_t)VQ_K * 4
                                  + (size_t)VQ_N * 4);

    vq_prep_kernel<<<VQ_K, 256, 0, stream>>>(emb, emb_hi, emb_lo, e2, loss_acc);
    vq_argmin_kernel<<<VQ_N / 128, 256, 0, stream>>>(z_e, emb_hi, emb_lo, e2,
                                                     idx, out_idx);
    vq_gather_loss_kernel<<<VQ_N, 256, 0, stream>>>(z_e, emb, idx, out_zq, loss_acc);
    vq_finalize_kernel<<<1, 1, 0, stream>>>(loss_acc, out_loss);
}
